// Attention_86569360818920
// MI455X (gfx1250) — matrix operand order repaired and checked by reading, not compiled, not run on hardware
//
#include <hip/hip_runtime.h>

typedef __attribute__((ext_vector_type(16))) _Float16 v16h;
typedef __attribute__((ext_vector_type(8)))  _Float16 v8h;
typedef __attribute__((ext_vector_type(8)))  float    v8f;

#define HID   1024
#define DKQ   64
#define DV    64
#define BATCH 8
#define SEQ   2048
#define NEG_INF (-1000000000.0f)

// ---------------------------------------------------------------------------
// Async memory->LDS copy (gfx1250): VDST = LDS byte address (VGPR),
// VADDR = 64-bit global address, SADDR = off.  Tracked by ASYNCcnt.
// ---------------------------------------------------------------------------
__device__ __forceinline__ void async_load_b128(void* lds_dst, const void* gsrc)
{
    const unsigned laddr = (unsigned)(unsigned long long)lds_dst;  // AS3 byte addr
    asm volatile("global_load_async_to_lds_b128 %0, %1, off"
                 :: "v"(laddr), "v"(gsrc)
                 : "memory");
}
__device__ __forceinline__ void wait_async0()
{
    asm volatile("s_wait_asynccnt 0x0" ::: "memory");
}

// ---------------------------------------------------------------------------
// W transpose + f16 convert: Wt[c][k] = f16(W[k][c]).
// ---------------------------------------------------------------------------
__global__ __launch_bounds__(256) void wt_f16_kernel(
    const float* __restrict__ W, _Float16* __restrict__ Wt, int hid, int dout)
{
    const int idx = blockIdx.x * 256 + threadIdx.x;
    if (idx < hid * dout) {
        const int k = idx / dout, c = idx % dout;
        Wt[(size_t)c * hid + k] = (_Float16)W[idx];
    }
}

// ---------------------------------------------------------------------------
// Projection: Y[r,c] = f16( (X[r,:] @ W[:,c] + bias[c]) * scale )
// One wave per 16x16 tile; K=1024 -> 32x v_wmma_f32_16x16x32_f16.
// ---------------------------------------------------------------------------
__global__ __launch_bounds__(128) void proj_f16_kernel(
    const float* __restrict__ X, const _Float16* __restrict__ Wt,
    const float* __restrict__ bias, _Float16* __restrict__ Y,
    int rows, int hid, int dout, float scale)
{
    const int wave = threadIdx.x >> 5;
    const int lane = threadIdx.x & 31;
    const int half = lane >> 4;
    const int rrow = lane & 15;

    const int numColTiles = dout >> 4;
    const int tile    = blockIdx.x * 4 + wave;
    const int rowTile = tile / numColTiles;
    const int colTile = tile % numColTiles;
    if (rowTile * 16 >= rows) return;            // wave-uniform

    const int arow = rowTile * 16 + rrow;        // A-matrix M for this lane
    const int bcol = colTile * 16 + rrow;        // B-matrix N for this lane

    v8f acc = {};
    for (int k0 = 0; k0 < hid; k0 += 32) {
        const float* xrow = X + (size_t)arow * hid + k0;
        if (k0 + 32 < hid) __builtin_prefetch(xrow + 32, 0, 1);
        // A 16x32: lane<16 -> K 0..15 ; lane>=16 -> K 16..31
        const float4 xa = *(const float4*)(xrow + half * 16);
        const float4 xb = *(const float4*)(xrow + half * 16 + 4);
        const float4 xc = *(const float4*)(xrow + half * 16 + 8);
        const float4 xd = *(const float4*)(xrow + half * 16 + 12);
        v16h a;
        a[0]=(_Float16)xa.x;  a[1]=(_Float16)xa.y;  a[2]=(_Float16)xa.z;  a[3]=(_Float16)xa.w;
        a[4]=(_Float16)xb.x;  a[5]=(_Float16)xb.y;  a[6]=(_Float16)xb.z;  a[7]=(_Float16)xb.w;
        a[8]=(_Float16)xc.x;  a[9]=(_Float16)xc.y;  a[10]=(_Float16)xc.z; a[11]=(_Float16)xc.w;
        a[12]=(_Float16)xd.x; a[13]=(_Float16)xd.y; a[14]=(_Float16)xd.z; a[15]=(_Float16)xd.w;
        // B 32x16: lane<16 -> K 0..15 ; lane>=16 -> K 16..31 ; contiguous in Wt
        const v16h b = *(const v16h*)(Wt + (size_t)bcol * hid + k0 + half * 16);
        acc = __builtin_amdgcn_wmma_f32_16x16x32_f16(false, a, false, b,
                                                     (short)0, acc, false, false);
    }
    const float bb = bias[bcol];
    #pragma unroll
    for (int j = 0; j < 8; ++j) {
        const int m = rowTile * 16 + j + 8 * half;    // C layout: M = j + 8*half
        Y[(size_t)m * dout + bcol] = (_Float16)((acc[j] + bb) * scale);
    }
}

// ---------------------------------------------------------------------------
// Flash attention: block = (batch, 64 q rows), wave = 16 q rows.
// K + mask tiles: double-buffered async global->LDS DMA (ASYNCcnt).
// V tile: register bounce with transposed LDS store (contiguous B-fragments).
// ---------------------------------------------------------------------------
__global__ __launch_bounds__(128) void flash_attn_kernel(
    const _Float16* __restrict__ Qf, const _Float16* __restrict__ Kf,
    const _Float16* __restrict__ Vf, const int* __restrict__ mask,
    float* __restrict__ out)
{
    __shared__ _Float16 lds_k[2][32 * DKQ];   // ping-pong K tiles   (8 KB)
    __shared__ int      lds_m[2][64 * 32];    // ping-pong mask tiles (16 KB)
    __shared__ _Float16 lds_vt[DV * 32];      // V transposed [vcol][key] (4 KB)
    __shared__ _Float16 lds_p[4][16 * 32];    // P bounce, per wave (4 KB)

    const int tid  = threadIdx.x;
    const int wave = tid >> 5;
    const int lane = tid & 31;
    const int half = lane >> 4;
    const int rrow = lane & 15;

    const int blocksPerBatch = SEQ / 64;
    const int b     = blockIdx.x / blocksPerBatch;
    const int qb0   = (blockIdx.x % blocksPerBatch) * 64;
    const int qbase = qb0 + wave * 16;

    // Q fragments (A layout); 1/sqrt(dk) folded into Q projection
    v16h aq0, aq1;
    {
        const v8h* qv = (const v8h*)(Qf + ((size_t)b * SEQ + qbase + rrow) * DKQ);
        const v8h q0 = qv[2 * half], q1 = qv[2 * half + 1], q2 = qv[4 + 2 * half], q3 = qv[5 + 2 * half];
        #pragma unroll
        for (int i = 0; i < 8; ++i) {
            aq0[i] = q0[i]; aq0[8 + i] = q1[i];
            aq1[i] = q2[i]; aq1[8 + i] = q3[i];
        }
    }

    v8f   o[4] = {};
    float m_i[8], l_i[8];
    #pragma unroll
    for (int j = 0; j < 8; ++j) { m_i[j] = -3.0e38f; l_i[j] = 0.0f; }

    const uint4* gk = (const uint4*)(Kf + (size_t)b * SEQ * DKQ);  // 8 uint4 / row
    const uint4* gv = (const uint4*)(Vf + (size_t)b * SEQ * DV);
    const int*   gm = mask + (size_t)b * SEQ * SEQ;

    uint4 rv[2];
    auto issue_async = [&](int k0, int buf) {
        uint4* sk4 = (uint4*)lds_k[buf];
        int4*  sm4 = (int4*)lds_m[buf];
        #pragma unroll
        for (int j = 0; j < 2; ++j)
            async_load_b128(sk4 + tid + j * 128, gk + (size_t)k0 * 8 + tid + j * 128);
        #pragma unroll
        for (int j = 0; j < 4; ++j) {
            const int idx = tid + j * 128;          // 512 int4 of 64x32 int tile
            const int qr  = idx >> 3;               // 8 int4 per row
            const int c4  = idx & 7;
            async_load_b128(sm4 + idx, gm + (size_t)(qb0 + qr) * SEQ + k0 + c4 * 4);
        }
    };
    auto load_v = [&](int k0) {
        #pragma unroll
        for (int j = 0; j < 2; ++j)
            rv[j] = gv[(size_t)k0 * 8 + tid + j * 128];
    };

    issue_async(0, 0);
    load_v(0);

    int buf = 0;
    for (int k0 = 0; k0 < SEQ; k0 += 32, buf ^= 1) {
        __syncthreads();                  // all waves done reading lds_vt / lds_p
        {   // commit V regs transposed: [vcol][key]
            #pragma unroll
            for (int j = 0; j < 2; ++j) {
                const int idx = tid + j * 128;
                const int key = idx >> 3;
                const int c8  = (idx & 7) * 8;
                union { uint4 u; _Float16 h[8]; } cv; cv.u = rv[j];
                #pragma unroll
                for (int e = 0; e < 8; ++e)
                    lds_vt[(c8 + e) * 32 + key] = cv.h[e];
            }
        }
        wait_async0();                    // this wave's K/mask DMA for 'buf' landed
        __syncthreads();                  // all waves' LDS writes visible

        if (k0 + 32 < SEQ) {              // overlap next tile's DMA with compute
            issue_async(k0 + 32, buf ^ 1);
            load_v(k0 + 32);
        }
        if (k0 + 64 < SEQ) {              // global_prefetch_b8, one more tile ahead
            __builtin_prefetch(gm + (size_t)(qb0 + (tid & 63)) * SEQ + (k0 + 64), 0, 1);
            __builtin_prefetch((const _Float16*)gk + (size_t)(k0 + 64) * DKQ + tid * 16, 0, 1);
            __builtin_prefetch((const _Float16*)gv + (size_t)(k0 + 64) * DV  + tid * 16, 0, 1);
        }

        // ---- scores for two 16x16 key subtiles ----
        float p[2][8], m_new[8];
        #pragma unroll
        for (int j = 0; j < 8; ++j) m_new[j] = m_i[j];

        #pragma unroll
        for (int s = 0; s < 2; ++s) {
            const _Float16* krow = lds_k[buf] + (s * 16 + rrow) * DKQ;  // B col N = key
            const v16h bk0 = *(const v16h*)(krow + half * 16);          // dims 0..31
            const v16h bk1 = *(const v16h*)(krow + 32 + half * 16);     // dims 32..63
            v8f c = {};
            c = __builtin_amdgcn_wmma_f32_16x16x32_f16(false, aq0, false, bk0,
                                                       (short)0, c, false, false);
            c = __builtin_amdgcn_wmma_f32_16x16x32_f16(false, aq1, false, bk1,
                                                       (short)0, c, false, false);
            #pragma unroll
            for (int j = 0; j < 8; ++j) {
                const int qr = wave * 16 + j + 8 * half;                // block-local q
                const int mv = lds_m[buf][qr * 32 + s * 16 + rrow];
                const float sc = mv ? c[j] : NEG_INF;
                p[s][j] = sc;
                m_new[j] = fmaxf(m_new[j], sc);
            }
        }
        // row max over the 16 lanes of each half-wave (N dimension)
        #pragma unroll
        for (int off = 1; off < 16; off <<= 1)
            #pragma unroll
            for (int j = 0; j < 8; ++j)
                m_new[j] = fmaxf(m_new[j], __shfl_xor(m_new[j], off, 16));

        float rs[8];
        #pragma unroll
        for (int j = 0; j < 8; ++j) {
            p[0][j] = __expf(p[0][j] - m_new[j]);
            p[1][j] = __expf(p[1][j] - m_new[j]);
            rs[j]   = p[0][j] + p[1][j];
        }
        #pragma unroll
        for (int off = 1; off < 16; off <<= 1)
            #pragma unroll
            for (int j = 0; j < 8; ++j)
                rs[j] += __shfl_xor(rs[j], off, 16);

        #pragma unroll
        for (int j = 0; j < 8; ++j) {
            const float sc = __expf(m_i[j] - m_new[j]);
            l_i[j] = l_i[j] * sc + rs[j];
            m_i[j] = m_new[j];
            #pragma unroll
            for (int t = 0; t < 4; ++t) o[t][j] *= sc;
        }

        // P: C-layout -> row-major 16x32 in LDS -> A-fragment
        #pragma unroll
        for (int s = 0; s < 2; ++s)
            #pragma unroll
            for (int j = 0; j < 8; ++j)
                lds_p[wave][(j + 8 * half) * 32 + s * 16 + rrow] = (_Float16)p[s][j];
        __syncthreads();

        v16h ap;
        {
            const v8h* pv = (const v8h*)(lds_p[wave] + rrow * 32);
            const v8h p0 = pv[2 * half], p1 = pv[2 * half + 1];
            #pragma unroll
            for (int i = 0; i < 8; ++i) { ap[i] = p0[i]; ap[8 + i] = p1[i]; }
        }
        // O += P @ Vtile; B-fragment contiguous thanks to transposed V
        #pragma unroll
        for (int t = 0; t < 4; ++t) {
            const v16h bvf = *(const v16h*)(lds_vt + (t * 16 + rrow) * 32 + half * 16);
            o[t] = __builtin_amdgcn_wmma_f32_16x16x32_f16(false, ap, false, bvf,
                                                          (short)0, o[t], false, false);
        }
    }

    // normalize and store (l_i > 0 always: masked rows are finite -1e9)
    #pragma unroll
    for (int j = 0; j < 8; ++j) {
        const float inv = 1.0f / l_i[j];
        const int q = qbase + j + 8 * half;
        #pragma unroll
        for (int t = 0; t < 4; ++t)
            out[((size_t)b * SEQ + q) * DV + t * 16 + rrow] = o[t][j] * inv;
    }
}

// ---------------------------------------------------------------------------
extern "C" void kernel_launch(void* const* d_in, const int* in_sizes, int n_in,
                              void* d_out, int out_size, void* d_ws, size_t ws_size,
                              hipStream_t stream)
{
    const float* k_hid = (const float*)d_in[0];
    const float* q_hid = (const float*)d_in[1];
    const float* v_hid = (const float*)d_in[2];
    const int*   mask  = (const int*)d_in[3];
    const float* Wq    = (const float*)d_in[4];
    const float* bq    = (const float*)d_in[5];
    const float* Wk    = (const float*)d_in[6];
    const float* bk    = (const float*)d_in[7];
    const float* Wv    = (const float*)d_in[8];
    const float* bv    = (const float*)d_in[9];

    const size_t NP = (size_t)BATCH * SEQ * DKQ;      // 1M halves per projection
    _Float16* qf  = (_Float16*)d_ws;
    _Float16* kf  = qf + NP;
    _Float16* vf  = kf + NP;
    _Float16* wtq = vf + NP;                          // 3 x 128 KB transposed weights
    _Float16* wtk = wtq + (size_t)HID * DKQ;
    _Float16* wtv = wtk + (size_t)HID * DKQ;

    const int nw = HID * DKQ;                         // 65536
    wt_f16_kernel<<<(nw + 255) / 256, 256, 0, stream>>>(Wq, wtq, HID, DKQ);
    wt_f16_kernel<<<(nw + 255) / 256, 256, 0, stream>>>(Wk, wtk, HID, DKQ);
    wt_f16_kernel<<<(nw + 255) / 256, 256, 0, stream>>>(Wv, wtv, HID, DV);

    const int rows      = BATCH * SEQ;                // 16384
    const int projWaves = (rows / 16) * (DKQ / 16);   // 4096
    dim3 pgrid(projWaves / 4), pblock(128);
    proj_f16_kernel<<<pgrid, pblock, 0, stream>>>(q_hid, wtq, bq, qf, rows, HID, DKQ, 0.125f);
    proj_f16_kernel<<<pgrid, pblock, 0, stream>>>(k_hid, wtk, bk, kf, rows, HID, DKQ, 1.0f);
    proj_f16_kernel<<<pgrid, pblock, 0, stream>>>(v_hid, wtv, bv, vf, rows, HID, DV, 1.0f);

    dim3 agrid(BATCH * SEQ / 64), ablock(128);
    flash_attn_kernel<<<agrid, ablock, 0, stream>>>(qf, kf, vf, mask, (float*)d_out);
}
